// NTXentLossChildrenPartwise_85624468013607
// MI455X (gfx1250) — matrix operand order repaired and checked by reading, not compiled, not run on hardware
//
#include <hip/hip_runtime.h>
#include <hip/hip_bf16.h>

#define Bv   32
#define Nv   1024
#define Dv   512
#define MBLK 128   // rows per workgroup (M blocking for L2 reuse)

typedef __attribute__((ext_vector_type(16))) __bf16       v16bf;
typedef __attribute__((ext_vector_type(8)))  float        v8f;
typedef __attribute__((ext_vector_type(4)))  unsigned int v4u;
typedef __attribute__((ext_vector_type(2)))  unsigned int v2u;
typedef __attribute__((ext_vector_type(4)))  float        v4f;

union V16U { v4u q[2]; v16bf v; };

__device__ __forceinline__ unsigned int f2bf(float f) {
    unsigned int u = __float_as_uint(f);
    u += 0x7fffu + ((u >> 16) & 1u);   // round-to-nearest-even
    return u >> 16;
}

// ---------------------------------------------------------------------------
// Kernel 1: L2-normalize rows, fold in 1/sqrt(T)=sqrt(2), convert to bf16.
// One wave per row (wave32), 16 floats per lane.
// ---------------------------------------------------------------------------
__global__ __launch_bounds__(256) void normalize_bf16_kernel(
    const float* __restrict__ ts, const float* __restrict__ rs,
    unsigned short* __restrict__ tn, unsigned short* __restrict__ rn)
{
    const int lane = threadIdx.x & 31;
    const int wib  = threadIdx.x >> 5;
    const long g   = (long)blockIdx.x * 8 + wib;          // 0 .. 2*B*N-1
    if (g >= 2L * Bv * Nv) return;
    const bool isT = (g < (long)Bv * Nv);
    const long row = isT ? g : g - (long)Bv * Nv;
    const float* src          = (isT ? ts : rs) + row * Dv;
    unsigned short* dst       = (isT ? tn : rn) + row * Dv;

    v4f x[4];
    float ss = 0.f;
#pragma unroll
    for (int c = 0; c < 4; ++c) {
        x[c] = ((const v4f*)src)[c * 32 + lane];
        ss += x[c].x * x[c].x + x[c].y * x[c].y + x[c].z * x[c].z + x[c].w * x[c].w;
    }
#pragma unroll
    for (int m = 16; m >= 1; m >>= 1) ss += __shfl_xor(ss, m, 32);
    const float scale = 1.41421356237f / fmaxf(sqrtf(ss), 1e-8f);

#pragma unroll
    for (int c = 0; c < 4; ++c) {
        v2u p;
        p.x = f2bf(x[c].x * scale) | (f2bf(x[c].y * scale) << 16);
        p.y = f2bf(x[c].z * scale) | (f2bf(x[c].w * scale) << 16);
        ((v2u*)dst)[c * 32 + lane] = p;
    }
}

// ---------------------------------------------------------------------------
// Kernel 2: fused GEMM (v_wmma_f32_16x16x32_bf16) + streaming logsumexp.
// Grid: B * (N/MBLK) workgroups of 4 waves; each wave owns 32 rows
// (two 16x16 C tiles) and iterates all 128 column tiles (tr then tt).
// A rows staged in LDS (128 KB dynamic), B operand from L2-resident global.
// ---------------------------------------------------------------------------
__global__ __launch_bounds__(128) void ntxent_main_kernel(
    const unsigned short* __restrict__ tn,
    const unsigned short* __restrict__ rn,
    float* __restrict__ partials)
{
    extern __shared__ unsigned short smemA[];           // MBLK * Dv bf16
    const int tid  = threadIdx.x;
    const int lane = tid & 31;
    const int wave = tid >> 5;
    const int b    = blockIdx.x >> 3;                   // N/MBLK == 8
    const int mb   = blockIdx.x & 7;
    const long batchOfs = (long)b * Nv * Dv;
    const int R0   = mb * MBLK;

    // cooperative stage of A rows [R0, R0+MBLK) into LDS
    {
        const v4u* gsrc = (const v4u*)(tn + batchOfs + (long)R0 * Dv);
        v4u* ldst = (v4u*)smemA;
#pragma unroll 4
        for (int i = tid; i < (MBLK * Dv) / 8; i += 128) ldst[i] = gsrc[i];
    }
    __syncthreads();

    const int half = (lane >> 4) & 1;
    const int l15  = lane & 15;
    const int w0l  = wave * 32;                          // local row base
    const unsigned short* a0row = smemA + (w0l + l15) * Dv;
    const unsigned short* a1row = smemA + (w0l + 16 + l15) * Dv;
    const int aK = half * 8;                             // A K-offset per half

    const unsigned short* tB = tn + batchOfs;
    const unsigned short* rB = rn + batchOfs;

    float s0[8], s1[8];
#pragma unroll
    for (int r = 0; r < 8; ++r) { s0[r] = 0.f; s1[r] = 0.f; }
    float posacc = 0.f;

    const int diag0 = (R0 + w0l) >> 4;                   // tr tile w/ tile0 diag
    const int diag1 = diag0 + 1;

    for (int t = 0; t < 128; ++t) {
        const unsigned short* bsrc = (t < 64)
            ? (rB + (long)(t << 4) * Dv)
            : (tB + (long)((t - 64) << 4) * Dv);
        const unsigned short* brow = bsrc + (long)l15 * Dv + half * 8;
        if (t + 1 < 128) {                               // prefetch next B tile
            const unsigned short* nsrc = (t + 1 < 64)
                ? (rB + (long)((t + 1) << 4) * Dv)
                : (tB + (long)((t + 1 - 64) << 4) * Dv);
            __builtin_prefetch(nsrc + (long)l15 * Dv, 0, 1);
        }

        v8f c0 = {}; v8f c1 = {};
#pragma unroll 4
        for (int k0 = 0; k0 < Dv; k0 += 32) {
            V16U a0u, a1u, bu;
            a0u.q[0] = *(const v4u*)(a0row + k0 + aK);
            a0u.q[1] = *(const v4u*)(a0row + k0 + aK + 16);
            a1u.q[0] = *(const v4u*)(a1row + k0 + aK);
            a1u.q[1] = *(const v4u*)(a1row + k0 + aK + 16);
            bu.q[0]  = *(const v4u*)(brow + k0);
            bu.q[1]  = *(const v4u*)(brow + k0 + 16);
            c0 = __builtin_amdgcn_wmma_f32_16x16x32_bf16(
                     false, a0u.v, false, bu.v, (short)0, c0, false, false);
            c1 = __builtin_amdgcn_wmma_f32_16x16x32_bf16(
                     false, a1u.v, false, bu.v, (short)0, c1, false, false);
        }

        // positives: tr diagonal tiles (C slots: VGPR r @ lane r / lane 24+r)
        if (t == diag0) {
            if (lane < 8) posacc += c0[lane];
            else if (lane >= 24) posacc += c0[lane - 24];
        }
        if (t == diag1) {
            if (lane < 8) posacc += c1[lane];
            else if (lane >= 24) posacc += c1[lane - 24];
        }
        // self-similarity mask on tt diagonal tiles
        if (t == 64 + diag0) {
            if (lane < 8) c0[lane] = -1e30f;
            else if (lane >= 24) c0[lane - 24] = -1e30f;
        }
        if (t == 64 + diag1) {
            if (lane < 8) c1[lane] = -1e30f;
            else if (lane >= 24) c1[lane - 24] = -1e30f;
        }

        // fixed-max streaming sum-of-exp (logits <= 2 by construction)
#pragma unroll
        for (int r = 0; r < 8; ++r) {
            s0[r] += __expf(c0[r] - 2.0f);
            s1[r] += __expf(c1[r] - 2.0f);
        }
    }

    // merge the 16 column-partials within each half-wave
#pragma unroll
    for (int m = 8; m >= 1; m >>= 1) {
#pragma unroll
        for (int r = 0; r < 8; ++r) {
            s0[r] += __shfl_xor(s0[r], m, 32);
            s1[r] += __shfl_xor(s1[r], m, 32);
        }
    }
    // per-half row-lse sum: lse = 2 + log(s); halves hold disjoint row sets
    float lsum = 0.f;
#pragma unroll
    for (int r = 0; r < 8; ++r) lsum += __logf(s0[r]) + __logf(s1[r]);
    float tot = lsum + __shfl_xor(lsum, 16, 32) + 64.0f;  // +2 per each of 32 rows
#pragma unroll
    for (int m = 16; m >= 1; m >>= 1) posacc += __shfl_xor(posacc, m, 32);

    if (lane == 0) partials[blockIdx.x * 4 + wave] = tot - posacc;
}

// ---------------------------------------------------------------------------
// Kernel 3: deterministic fixed-tree reduction of 1024 partials -> scalar.
// ---------------------------------------------------------------------------
__global__ __launch_bounds__(256) void finalize_kernel(
    const float* __restrict__ partials, float* __restrict__ out)
{
    __shared__ float red[256];
    const int tid = threadIdx.x;
    red[tid] = partials[tid] + partials[tid + 256] +
               partials[tid + 512] + partials[tid + 768];
    __syncthreads();
    for (int s = 128; s > 0; s >>= 1) {
        if (tid < s) red[tid] += red[tid + s];
        __syncthreads();
    }
    if (tid == 0) out[0] = red[0] / 64.0f;   // / (2 * BATCH_SIZE)
}

// ---------------------------------------------------------------------------
extern "C" void kernel_launch(void* const* d_in, const int* in_sizes, int n_in,
                              void* d_out, int out_size, void* d_ws, size_t ws_size,
                              hipStream_t stream)
{
    const float* ts = (const float*)d_in[0];   // children_ts [B,N,D] f32
    const float* rs = (const float*)d_in[1];   // children_rs [B,N,D] f32
    // d_in[2] (partnet_ids) is unused by the reference loss.

    const size_t elems = (size_t)Bv * Nv * Dv;
    unsigned short* tn = (unsigned short*)d_ws;            // bf16, scaled
    unsigned short* rn = tn + elems;                       // bf16, scaled
    float* partials    = (float*)(rn + elems);             // 1024 floats
    float* out         = (float*)d_out;

    normalize_bf16_kernel<<<(2 * Bv * Nv) / 8, 256, 0, stream>>>(ts, rs, tn, rn);
    ntxent_main_kernel<<<Bv * (Nv / MBLK), 128, (size_t)MBLK * Dv * 2, stream>>>(
        tn, rn, partials);
    finalize_kernel<<<1, 256, 0, stream>>>(partials, out);
}
